// TemporalAttention_39307540693353
// MI455X (gfx1250) — matrix operand order repaired and checked by reading, not compiled, not run on hardware
//
#include <hip/hip_runtime.h>

typedef __attribute__((ext_vector_type(16))) _Float16 v16h;
typedef __attribute__((ext_vector_type(8)))  float    v8f;
typedef __attribute__((ext_vector_type(4)))  float    v4f;

#define BLOCK  256
#define WAVES  8      // waves per block (wave32)
#define TPW    16     // t-rows per wave (WMMA M)
#define NT     2048   // n-tile staged in LDS
#define NSPLIT 8      // grid.y split over the N dimension

static constexpr float LOG2E = 1.4426950408889634f;

// ---------------------------------------------------------------------------
// Kernel 1: a2[n] = -0.5*log2e * sum_p ((param[n,p]-norm[p])/sigma[p])^2
//           also zeroes the num/den accumulators (graph-replay safe).
// ---------------------------------------------------------------------------
__global__ void ta_prep(const float* __restrict__ param_mat,
                        const float* __restrict__ target_norm,
                        const float* __restrict__ param_sigma,
                        float* __restrict__ a2,
                        float* __restrict__ num,
                        float* __restrict__ den,
                        int N, int T) {
  int n = blockIdx.x * blockDim.x + threadIdx.x;
  if (n < N) {
    v4f pm = *(const v4f*)(param_mat + 4 * n);
    float s = 0.0f;
#pragma unroll
    for (int p = 0; p < 4; ++p) {
      float d = (pm[p] - target_norm[p]) / param_sigma[p];
      s = fmaf(d, d, s);
    }
    a2[n] = -0.5f * LOG2E * s;
  }
  if (n < T) { num[n] = 0.0f; den[n] = 0.0f; }
}

// ---------------------------------------------------------------------------
// Kernel 2: streaming fused softmax-matvec.
// Each wave owns 16 t-rows; per 32-n chunk it computes w = exp(logit) in f32,
// packs into the WMMA A layout (f16), and accumulates
//   D[:,0] += w @ th     D[:,1] += w @ 1
// with v_wmma_f32_16x16x32_f16.  Partial sums combined via f32 atomics.
// ---------------------------------------------------------------------------
__global__ __launch_bounds__(BLOCK) void ta_main(
    const float* __restrict__ target_t,   // [T]
    const float* __restrict__ t_mat,      // [N]
    const float* __restrict__ th_mat,     // [N]
    const float* __restrict__ a2,         // [N]
    float* __restrict__ num,
    float* __restrict__ den,
    int N, int T) {
  __shared__ float s_tm[NT];
  __shared__ float s_a2[NT];
  __shared__ float s_th[NT];

  const int tid  = threadIdx.x;
  const int wave = tid >> 5;
  const int lane = tid & 31;
  const int m    = lane & 15;                // A-matrix row for this lane
  const int kb   = (lane < 16) ? 0 : 8;      // A-layout K base for half-wave

  const int tbase = blockIdx.x * (WAVES * TPW) + wave * TPW;
  int ti = tbase + m; if (ti >= T) ti = T - 1;
  const float tt = target_t[ti];
  const float Ct = -50.0f * LOG2E;           // -0.5/sigma_t^2 * log2(e)

  const int nper   = N / gridDim.y;          // n-range per block
  const int nstart = blockIdx.y * nper;

  v8f acc = {};

  for (int tile = 0; tile < nper; tile += NT) {
    const int nb = nstart + tile;
    __syncthreads();
    for (int i = tid; i < NT; i += BLOCK) {
      s_tm[i] = t_mat[nb + i];
      s_a2[i] = a2[nb + i];
      s_th[i] = th_mat[nb + i];
      if (tile + NT < nper) {                // prefetch next tile (global_prefetch_b8)
        __builtin_prefetch(&t_mat[nb + NT + i], 0, 1);
        __builtin_prefetch(&a2[nb + NT + i], 0, 1);
      }
    }
    __syncthreads();

#pragma unroll 2
    for (int c = 0; c < NT; c += 32) {
      // Build A (16x32 f16): elements 0..7 -> K = kb+0..7,
      //                      elements 8..15 -> K = kb+16..23
      v16h a;
#pragma unroll
      for (int i = 0; i < 8; ++i) {
        float d0 = s_tm[c + kb + i] - tt;
        float w0 = __builtin_amdgcn_exp2f(fmaf(d0 * d0, Ct, s_a2[c + kb + i]));
        a[i] = (_Float16)w0;
        float d1 = s_tm[c + kb + 16 + i] - tt;
        float w1 = __builtin_amdgcn_exp2f(fmaf(d1 * d1, Ct, s_a2[c + kb + 16 + i]));
        a[8 + i] = (_Float16)w1;
      }
      // Build B (32x16 f16): lane%16 == N, elements in same K order as A; column 0 = th, column 1 = 1.0
      v16h b = {};
#pragma unroll
      for (int i = 0; i < 8; ++i) {
        if (m == 0) {
          b[i] = (_Float16)s_th[c + kb + i];
          b[8 + i] = (_Float16)s_th[c + kb + 16 + i];
        } else if (m == 1) {
          b[i] = (_Float16)1.0f;
          b[8 + i] = (_Float16)1.0f;
        }
      }

      acc = __builtin_amdgcn_wmma_f32_16x16x32_f16(
          /*neg_a=*/false, a, /*neg_b=*/false, b,
          /*c_mod=*/(short)0, acc, /*reuse_a=*/false, /*reuse_b=*/false);
    }
  }

  // D layout: VGPR r, lane<16 -> (M=r, N=lane); lane>=16 -> (M=r+8, N=lane-16)
  const int col  = lane & 15;
  const int roff = (lane < 16) ? 0 : 8;
  if (col == 0) {
#pragma unroll
    for (int r = 0; r < 8; ++r) {
      int t = tbase + roff + r;
      if (t < T) unsafeAtomicAdd(&num[t], acc[r]);
    }
  } else if (col == 1) {
#pragma unroll
    for (int r = 0; r < 8; ++r) {
      int t = tbase + roff + r;
      if (t < T) unsafeAtomicAdd(&den[t], acc[r]);
    }
  }
}

// ---------------------------------------------------------------------------
// Kernel 3: thickness[t] = num[t] / den[t]
// ---------------------------------------------------------------------------
__global__ void ta_finalize(const float* __restrict__ num,
                            const float* __restrict__ den,
                            float* __restrict__ out, int T) {
  int t = blockIdx.x * blockDim.x + threadIdx.x;
  if (t < T) out[t] = num[t] / den[t];
}

extern "C" void kernel_launch(void* const* d_in, const int* in_sizes, int n_in,
                              void* d_out, int out_size, void* d_ws, size_t ws_size,
                              hipStream_t stream) {
  const float* target_t    = (const float*)d_in[0];  // [T,1]
  const float* param_mat   = (const float*)d_in[1];  // [N,4]
  const float* t_mat       = (const float*)d_in[2];  // [N]
  const float* th_mat      = (const float*)d_in[3];  // [N]
  const float* target_norm = (const float*)d_in[4];  // [4]
  const float* param_sigma = (const float*)d_in[5];  // [4]
  const int T = in_sizes[0];
  const int N = in_sizes[2];

  float* a2  = (float*)d_ws;      // N floats
  float* num = a2 + N;            // T floats
  float* den = num + T;           // T floats
  float* out = (float*)d_out;     // [1,T] float32

  ta_prep<<<(N + BLOCK - 1) / BLOCK, BLOCK, 0, stream>>>(
      param_mat, target_norm, param_sigma, a2, num, den, N, T);

  dim3 grid((T + WAVES * TPW - 1) / (WAVES * TPW), NSPLIT);
  ta_main<<<grid, BLOCK, 0, stream>>>(target_t, t_mat, th_mat, a2, num, den, N, T);

  ta_finalize<<<(T + BLOCK - 1) / BLOCK, BLOCK, 0, stream>>>(num, den, out, T);
}
